// MultiColonyModel_43224550867195
// MI455X (gfx1250) — matrix operand order repaired and checked by reading, not compiled, not run on hardware
//
#include <hip/hip_runtime.h>
#include <math.h>

// CDNA5 (gfx1250) wave32 WMMA types
typedef __attribute__((ext_vector_type(16))) _Float16 v16h;
typedef __attribute__((ext_vector_type(8)))  float    v8f;
typedef __attribute__((ext_vector_type(2)))  _Float16 v2h;

#define WAVES_PER_BLOCK 4
#define MAX_BLOCKS 1024

// Per-wave LDS scratch
struct WaveLds {
  float    xs[16 * 16];       // x tile f32 (cols 14,15 kept at exact 0)
  _Float16 st16[7][16 * 16];  // colony states f16 (cols 14,15 exact 0)
  _Float16 cw16[16 * 128];    // coupled states, flat [row][p = c*14 + o]
};

// A-matrix (16x32 f16) element K for (vgpr j, half t, lane-half h)
__device__ __forceinline__ int a_k(int j, int t, int h) {
  return ((j >> 2) << 4) + (h << 3) + ((j & 3) << 1) + t;
}
// B-matrix (32x16 f16) element K for (vgpr j, half t, lane-half h)
__device__ __forceinline__ int b_k(int j, int t, int h) {
  return ((j >> 2) << 4) + (h << 3) + ((j & 3) << 1) + t;
}

__device__ __forceinline__ v2h splat2(float f) {
  _Float16 h = (_Float16)f;
  v2h r = {h, h};
  return r;
}
__device__ __forceinline__ v2h pack2(float a, float b) {
  v2h r;                       // lowers to v_cvt_pk_f16_f32
  r[0] = (_Float16)a;
  r[1] = (_Float16)b;
  return r;
}

// packed f16 hardware tanh (true16 halves); earlyclobber since dest is written
// before src is fully consumed; v_nop covers the TRANS result-use hazard
__device__ __forceinline__ v2h tanh_pk(v2h x) {
  v2h r;
  asm volatile("v_tanh_f16 %0.l, %1.l\n\t"
               "v_tanh_f16 %0.h, %1.h\n\t"
               "v_nop"
               : "=&v"(r) : "v"(x));
  return r;
}
// packed branch-free tanh-GELU: 0.5x(1+tanh(x(C1+C2*x^2)))
__device__ __forceinline__ v2h gelu_pk(v2h x) {
  const v2h C1 = splat2(0.7978845608f);
  const v2h C2 = splat2(0.035677408f);   // 0.7978845608 * 0.044715
  const v2h H  = splat2(0.5f);
  v2h x2 = x * x;
  v2h w  = x2 * C2 + C1;                 // v_pk_fma_f16
  v2h u  = x * w;
  v2h t  = tanh_pk(u);
  v2h hx = x * H;
  return hx * t + hx;                    // v_pk_fma_f16
}

__global__ __launch_bounds__(WAVES_PER_BLOCK * 32)
void fano_colony_kernel(const float* __restrict__ x,
                        const float* __restrict__ Wc,
                        const float* __restrict__ bc,
                        const float* __restrict__ gate_c,
                        const float* __restrict__ Wline,
                        const float* __restrict__ bline,
                        const float* __restrict__ gates_f,
                        const float* __restrict__ Wout,
                        const float* __restrict__ bout,
                        float* __restrict__ out, int Bn)
{
  __shared__ WaveLds smem[WAVES_PER_BLOCK];
  const int lane = threadIdx.x & 31;
  const int wv   = threadIdx.x >> 5;
  WaveLds& W = smem[wv];
  const int Ml = lane & 15;   // M for A-frags / N for B,D frags
  const int hl = lane >> 4;   // lane half

  // Fano plane lines (i, j) -> k
  const int li[7] = {0, 0, 0, 1, 4, 3, 4};
  const int lj[7] = {1, 3, 5, 3, 1, 2, 2};
  const int lk[7] = {2, 4, 6, 5, 6, 6, 5};

  // ---------- per-lane packed constants ----------
  v2h sg2[7], gl2[7];
  #pragma unroll
  for (int c = 0; c < 7; ++c)
    sg2[c] = splat2(1.f / (1.f + __expf(-gate_c[c])));
  {   // softmax over the 7 line gates
    float gmax = gates_f[0];
    #pragma unroll
    for (int c = 1; c < 7; ++c) gmax = fmaxf(gmax, gates_f[c]);
    float ge[7], gsum = 0.f;
    #pragma unroll
    for (int c = 0; c < 7; ++c) { ge[c] = __expf(gates_f[c] - gmax); gsum += ge[c]; }
    #pragma unroll
    for (int c = 0; c < 7; ++c) gl2[c] = splat2(ge[c] / gsum);
  }

  // ---------- persistent weight B-fragments (biases folded into spare K rows) ----------
  v16h Bc[7];                    // row k<14: Wc^T ; row k==14: bc (pairs with A's 1.0)
  #pragma unroll
  for (int c = 0; c < 7; ++c) {
    v16h f;
    #pragma unroll
    for (int j = 0; j < 8; ++j)
      #pragma unroll
      for (int t = 0; t < 2; ++t) {
        int k = b_k(j, t, hl);
        float v = 0.f;
        if (Ml < 14) {
          if (k < 14)       v = Wc[c * 196 + Ml * 14 + k];
          else if (k == 14) v = bc[c * 14 + Ml];
        }
        f[2 * j + t] = (_Float16)v;
      }
    Bc[c] = f;
  }
  v16h Bl;                       // rows k<28: Wline^T ; row k==30: bline
  #pragma unroll
  for (int j = 0; j < 8; ++j)
    #pragma unroll
    for (int t = 0; t < 2; ++t) {
      int k = b_k(j, t, hl);
      float v = 0.f;
      if (Ml < 14) {
        if (k < 28)       v = Wline[Ml * 28 + k];
        else if (k == 30) v = bline[Ml];
      }
      Bl[2 * j + t] = (_Float16)v;
    }
  v16h Bo[4];                    // rows p<98: Wout^T ; row p==98: bout
  #pragma unroll
  for (int m = 0; m < 4; ++m) {
    v16h f;
    #pragma unroll
    for (int j = 0; j < 8; ++j)
      #pragma unroll
      for (int t = 0; t < 2; ++t) {
        int p = 32 * m + b_k(j, t, hl);
        float v = 0.f;
        if (Ml < 14) {
          if (p < 98)       v = Wout[Ml * 98 + p];
          else if (p == 98) v = bout[Ml];
        }
        f[2 * j + t] = (_Float16)v;
      }
    Bo[m] = f;
  }

  // zero the x-tile pad columns once (never touched again)
  #pragma unroll
  for (int idx = lane; idx < 256; idx += 32)
    if ((idx & 15) >= 14) W.xs[idx] = 0.f;
  asm volatile("s_wait_dscnt 0" ::: "memory");

  // ---------- grid-stride over 16-row batch tiles ----------
  const int tiles = (Bn + 15) >> 4;
  const int gw = blockIdx.x * WAVES_PER_BLOCK + wv;
  const int nw = gridDim.x * WAVES_PER_BLOCK;

  for (int tile = gw; tile < tiles; tile += nw) {
    const long long row0 = (long long)tile << 4;
    const bool fullTile = (row0 + 16 <= (long long)Bn);

    // stream x tile into LDS: lane owns column k=lane (k<14), rows 0..15
    if (lane < 14) {
      const float* xp = x + row0 * 14 + lane;
      if (fullTile) {
        #pragma unroll
        for (int r = 0; r < 16; ++r)
          W.xs[r * 16 + lane] = xp[r * 14];
      } else {
        #pragma unroll
        for (int r = 0; r < 16; ++r)
          W.xs[r * 16 + lane] = (row0 + r < (long long)Bn) ? xp[r * 14] : 0.f;
      }
    }
    asm volatile("s_wait_dscnt 0" ::: "memory");

    // A-fragment of x; K=14 carries the constant 1.0 that pairs with bias rows
    v16h Ax;
    #pragma unroll
    for (int j = 0; j < 8; ++j)
      #pragma unroll
      for (int t = 0; t < 2; ++t) {
        int k = a_k(j, t, hl);
        _Float16 v;
        if (k < 14)       v = (_Float16)W.xs[Ml * 16 + k];
        else if (k == 14) v = (_Float16)1.0f;
        else              v = (_Float16)0.f;
        Ax[2 * j + t] = v;
      }

    // residual x values, packed to f16 pairs (rows 2q, 2q+1)
    v2h xr2[4];
    #pragma unroll
    for (int q = 0; q < 4; ++q) {
      int r0 = 2 * q + (hl << 3);
      xr2[q] = pack2(W.xs[r0 * 16 + Ml], W.xs[(r0 + 1) * 16 + Ml]);
    }

    // -------- stage 1: s = x + sig(g)*gelu(x Wc^T + bc) --------
    v2h cp7[7][4];    // packed coupled accumulators (rows 2q,2q+1 per slot)
    #pragma unroll
    for (int c = 0; c < 7; ++c) {
      v8f z = {};
      v8f acc = __builtin_amdgcn_wmma_f32_16x16x32_f16(
          false, Ax, false, Bc[c], (short)0, z, false, false);
      #pragma unroll
      for (int q = 0; q < 4; ++q) {
        int r0 = 2 * q + (hl << 3);
        v2h a2 = pack2(acc[2 * q], acc[2 * q + 1]);
        v2h s2 = sg2[c] * gelu_pk(a2) + xr2[q];      // v_pk_fma_f16
        W.st16[c][r0 * 16 + Ml]       = s2[0];       // pad cols get exact 0
        W.st16[c][(r0 + 1) * 16 + Ml] = s2[1];
        cp7[c][q] = s2;
      }
    }
    asm volatile("s_wait_dscnt 0" ::: "memory");

    // -------- stage 2: Fano coupling (packed register accumulation) --------
    #pragma unroll
    for (int l = 0; l < 7; ++l) {
      v16h Al;
      #pragma unroll
      for (int j = 0; j < 8; ++j)
        #pragma unroll
        for (int t = 0; t < 2; ++t) {
          int k = a_k(j, t, hl);
          _Float16 v;
          if (k < 14)       v = W.st16[li[l]][Ml * 16 + k];
          else if (k < 30)  v = W.st16[lj[l]][Ml * 16 + (k - 14)]; // 28,29 hit zeros
          else if (k == 30) v = (_Float16)1.0f;   // pairs with bline row
          else              v = (_Float16)0.f;
          Al[2 * j + t] = v;
        }
      v8f z = {};
      v8f dacc = __builtin_amdgcn_wmma_f32_16x16x32_f16(
          false, Al, false, Bl, (short)0, z, false, false);
      #pragma unroll
      for (int q = 0; q < 4; ++q) {
        v2h d2 = pack2(dacc[2 * q], dacc[2 * q + 1]);
        cp7[lk[l]][q] = d2 * gl2[l] + cp7[lk[l]][q];  // v_pk_fma_f16
      }
    }

    // flush coupled buffer, ascending colony order, UNGUARDED:
    // pad-lane (Ml=14,15) writes land on colony c+1's o=0,1 and are overwritten
    // by the later colony; colony 6 pads land at p=98,99 (never read).
    #pragma unroll
    for (int c = 0; c < 7; ++c)
      #pragma unroll
      for (int q = 0; q < 4; ++q) {
        int r0 = 2 * q + (hl << 3);
        W.cw16[r0 * 128 + c * 14 + Ml]       = cp7[c][q][0];
        W.cw16[(r0 + 1) * 128 + c * 14 + Ml] = cp7[c][q][1];
      }
    asm volatile("s_wait_dscnt 0" ::: "memory");

    // -------- stage 3: out = coupled(98) @ Wout^T + bout --------
    v8f oacc = {};
    #pragma unroll
    for (int m = 0; m < 4; ++m) {
      v16h Am;
      #pragma unroll
      for (int j = 0; j < 8; ++j)
        #pragma unroll
        for (int t = 0; t < 2; ++t) {
          int p = 32 * m + a_k(j, t, hl);
          _Float16 v;
          if (p < 98)       v = W.cw16[Ml * 128 + p];
          else if (p == 98) v = (_Float16)1.0f;    // pairs with bout row
          else              v = (_Float16)0.f;
          Am[2 * j + t] = v;
        }
      oacc = __builtin_amdgcn_wmma_f32_16x16x32_f16(
          false, Am, false, Bo[m], (short)0, oacc, false, false);
    }
    if (fullTile) {
      if (Ml < 14) {
        float* op = out + (row0 + (hl << 3)) * 14 + Ml;
        #pragma unroll
        for (int j = 0; j < 8; ++j)
          op[j * 14] = oacc[j];
      }
    } else {
      #pragma unroll
      for (int j = 0; j < 8; ++j) {
        long long r = row0 + j + (hl << 3);
        if (Ml < 14 && r < (long long)Bn)
          out[r * 14 + Ml] = oacc[j];
      }
    }
  }
}

extern "C" void kernel_launch(void* const* d_in, const int* in_sizes, int n_in,
                              void* d_out, int out_size, void* d_ws, size_t ws_size,
                              hipStream_t stream) {
  const float* x       = (const float*)d_in[0];
  const float* Wc      = (const float*)d_in[1];
  const float* bc      = (const float*)d_in[2];
  const float* gate_c  = (const float*)d_in[3];
  const float* Wline   = (const float*)d_in[4];
  const float* bline   = (const float*)d_in[5];
  const float* gates_f = (const float*)d_in[6];
  const float* Wout    = (const float*)d_in[7];
  const float* bout    = (const float*)d_in[8];
  float* out = (float*)d_out;

  const int Bn = in_sizes[0] / 14;
  const int tiles = (Bn + 15) / 16;
  int blocks = (tiles + WAVES_PER_BLOCK - 1) / WAVES_PER_BLOCK;
  if (blocks > MAX_BLOCKS) blocks = MAX_BLOCKS;

  fano_colony_kernel<<<blocks, WAVES_PER_BLOCK * 32, 0, stream>>>(
      x, Wc, bc, gate_c, Wline, bline, gates_f, Wout, bout, out, Bn);
}
